// LSTMModule_57346403336522
// MI455X (gfx1250) — compile-verified, matrix operand order checked by reading, not run on hardware
//
#include <hip/hip_runtime.h>
#include <hip/hip_bf16.h>
#include <math.h>

typedef __bf16 bf16_t;
typedef __attribute__((ext_vector_type(16))) __bf16 v16bf;
typedef __attribute__((ext_vector_type(8)))  __bf16 v8bf;
typedef __attribute__((ext_vector_type(8)))  float  v8f;

#define NB   32
#define NS   256
#define ND   512
#define NH   512
#define NO   10000
#define G4H  2048
#define NWG  16     // workgroups in the persistent recurrent kernel
#define DEC_CT 625  // 10000/16 column tiles in decoder

__device__ __forceinline__ float sigf(float x)   { return 1.0f / (1.0f + __expf(-x)); }
__device__ __forceinline__ float tanhf_(float x) { return 1.0f - 2.0f / (__expf(2.0f * x) + 1.0f); }

__device__ __forceinline__ v8f wmma_bf16(v16bf a, v16bf b, v8f c) {
  return __builtin_amdgcn_wmma_f32_16x16x32_bf16(false, a, false, b, (short)0, c, false, false);
}

// ---------------------------------------------------------------------------
// Weight repack: f32 [4][K][512] gate weights -> bf16 WMMA B-fragments.
// Fragment storage (K-contiguous per column tile for immediate-offset loads):
//   dst[((ct*NKC + kc)*32 + lane)*16 + e]
// with n = lane&15, khalf = lane>>4, k_local = ((e>>3)<<4) + khalf*8 + (e&7)
// ---------------------------------------------------------------------------
__global__ __launch_bounds__(256) void k_repack_gate512(const float* __restrict__ src,
                                                        bf16_t* __restrict__ dst) {
  int tid = blockIdx.x * 256 + threadIdx.x;          // 128ct * 16kc * 512 = 1,048,576
  int ct  = tid >> 13;                               // / (16*512)
  int rem = tid & 8191;
  int kc  = rem >> 9;
  int li  = rem & 511;
  int lane = li >> 4, e = li & 15;
  int n = lane & 15, khalf = lane >> 4;
  int kl  = ((e >> 3) << 4) + khalf * 8 + (e & 7);
  int k   = kc * 32 + kl;
  int col = ct * 16 + n;
  int g = col >> 9, h = col & 511;
  dst[tid] = (bf16_t)src[((size_t)(g * 512 + k)) * 512 + h];
}

// Layer-1 combined [Wi1; Wh1] : K = 1024 (32 kc chunks per column tile)
__global__ __launch_bounds__(256) void k_repack_w1(const float* __restrict__ wi1,
                                                   const float* __restrict__ wh1,
                                                   bf16_t* __restrict__ dst) {
  int tid = blockIdx.x * 256 + threadIdx.x;          // 128ct * 32kc * 512 = 2,097,152
  int ct  = tid >> 14;                               // / (32*512)
  int rem = tid & 16383;
  int kc  = rem >> 9;
  int li  = rem & 511;
  int lane = li >> 4, e = li & 15;
  int n = lane & 15, khalf = lane >> 4;
  int kl  = ((e >> 3) << 4) + khalf * 8 + (e & 7);
  int k   = kc * 32 + kl;
  int col = ct * 16 + n;
  int g = col >> 9, h = col & 511;
  float v = (k < 512) ? wi1[((size_t)(g * 512 + k)) * 512 + h]
                      : wh1[((size_t)(g * 512 + (k - 512))) * 512 + h];
  dst[tid] = (bf16_t)v;
}

__global__ __launch_bounds__(256) void k_repack_wdec(const float* __restrict__ wdec,
                                                     bf16_t* __restrict__ dst) {
  int tid = blockIdx.x * 256 + threadIdx.x;          // 625ct * 16kc * 512 = 5,120,000
  int ct  = tid >> 13;                               // / (16*512)
  int rem = tid & 8191;
  int kc  = rem >> 9;
  int li  = rem & 511;
  int lane = li >> 4, e = li & 15;
  int n = lane & 15, khalf = lane >> 4;
  int kl  = ((e >> 3) << 4) + khalf * 8 + (e & 7);
  int k   = kc * 32 + kl;
  int col = ct * 16 + n;
  dst[tid] = (bf16_t)wdec[(size_t)k * NO + col];
}

__global__ __launch_bounds__(256) void k_bias1(const float* __restrict__ b_i,
                                               const float* __restrict__ b_h,
                                               float* __restrict__ bias1) {
  int tid = blockIdx.x * 256 + threadIdx.x;          // 2048
  bias1[tid] = b_i[(4 + (tid >> 9)) * 512 + (tid & 511)] +
               b_h[(4 + (tid >> 9)) * 512 + (tid & 511)];
}

// ---------------------------------------------------------------------------
// X0[s] = emb(ids[:,s]) @ W_i[0] + b_i[0] + b_h[0], stored "column-major per
// step": X0[((s*2048 + col)*32) + m]  so accumulator init = two float4 loads.
// grid = (16 colblocks, 256 steps), 256 threads (8 waves, 2 tiles/wave)
// ---------------------------------------------------------------------------
__global__ __launch_bounds__(256) void k_x0(const int* __restrict__ ids,
                                            const float* __restrict__ emb,
                                            const float* __restrict__ b_i,
                                            const float* __restrict__ b_h,
                                            const bf16_t* __restrict__ wi0f,
                                            float* __restrict__ X0) {
  __shared__ bf16_t Abuf[32][512];                   // emb rows for this step (bf16)
  const int w = blockIdx.x;                          // colblock 0..15
  const int s = blockIdx.y;
  const int tid = threadIdx.x;
  {
    const int b = tid >> 3;
    const int c0 = (tid & 7) * 64;
    const int id = ids[b * NS + s];
    const float* row = emb + (size_t)id * ND + c0;
    #pragma unroll
    for (int j = 0; j < 64; ++j) Abuf[b][c0 + j] = (bf16_t)row[j];
  }
  __syncthreads();

  const int lane = tid & 31, wv = tid >> 5;
  const int rt = wv & 1;
  const int ct0 = w * 8 + (wv >> 1) * 2;
  const int n = lane & 15, khalf = lane >> 4;

  v8f acc0, acc1;
  {
    int c0 = ct0 * 16 + n, c1 = c0 + 16;
    int g0 = c0 >> 9, h0 = c0 & 511, g1 = c1 >> 9, h1 = c1 & 511;
    float bb0 = b_i[g0 * 512 + h0] + b_h[g0 * 512 + h0];
    float bb1 = b_i[g1 * 512 + h1] + b_h[g1 * 512 + h1];
    #pragma unroll
    for (int r = 0; r < 8; ++r) { acc0[r] = bb0; acc1[r] = bb1; }
  }

  const int m = rt * 16 + n;
  const v16bf* bp0 = ((const v16bf*)wi0f) + (size_t)(ct0)     * 16 * 32 + lane;
  const v16bf* bp1 = ((const v16bf*)wi0f) + (size_t)(ct0 + 1) * 16 * 32 + lane;
  #pragma unroll
  for (int kc = 0; kc < 16; ++kc) {
    int k0 = kc * 32 + khalf * 8;
    v8bf lo = *(const v8bf*)&Abuf[m][k0];
    v8bf hi = *(const v8bf*)&Abuf[m][k0 + 16];
    v16bf a;
    #pragma unroll
    for (int i = 0; i < 8; ++i) { a[i] = lo[i]; a[i + 8] = hi[i]; }
    v16bf b0 = bp0[kc * 32];
    v16bf b1 = bp1[kc * 32];
    acc0 = wmma_bf16(a, b0, acc0);
    acc1 = wmma_bf16(a, b1, acc1);
  }

  const int mb = rt * 16 + khalf * 8;
  const int c0 = ct0 * 16 + n;
  float* p0 = X0 + ((size_t)s * G4H + c0) * 32 + mb;
  float* p1 = p0 + 16 * 32;
  *(float4*)(p0)     = make_float4(acc0[0], acc0[1], acc0[2], acc0[3]);
  *(float4*)(p0 + 4) = make_float4(acc0[4], acc0[5], acc0[6], acc0[7]);
  *(float4*)(p1)     = make_float4(acc1[0], acc1[1], acc1[2], acc1[3]);
  *(float4*)(p1 + 4) = make_float4(acc1[4], acc1[5], acc1[6], acc1[7]);
}

// ---------------------------------------------------------------------------
// Global barrier: monotonically increasing target; cnt/phase zeroed at launch.
// ---------------------------------------------------------------------------
__device__ __forceinline__ void grid_barrier(int* cnt, int* phase, int target) {
  __syncthreads();
  if (threadIdx.x == 0) {
    __threadfence();
    int arrived = __hip_atomic_fetch_add(cnt, 1, __ATOMIC_ACQ_REL,
                                         __HIP_MEMORY_SCOPE_AGENT) + 1;
    if (arrived == target * NWG) {
      __hip_atomic_store(phase, target, __ATOMIC_RELEASE, __HIP_MEMORY_SCOPE_AGENT);
    } else {
      while (__hip_atomic_load(phase, __ATOMIC_ACQUIRE, __HIP_MEMORY_SCOPE_AGENT) < target)
        __builtin_amdgcn_s_sleep(2);
    }
  }
  __syncthreads();
}

// ---------------------------------------------------------------------------
// Persistent recurrent kernel: 16 WGs x 256 threads (8 waves).
// WG w owns hidden units [32w, 32w+32) for both layers; wave (rt,g) computes
// gate g, batch rowtile rt, two 16x16 tiles (units 32w..+15, 32w+16..+31).
// Weight fragments are K-contiguous per column tile -> immediate offsets.
// ---------------------------------------------------------------------------
__global__ __launch_bounds__(256, 1) void k_lstm(const float* __restrict__ X0,
                                                 const bf16_t* __restrict__ wh0f,
                                                 const bf16_t* __restrict__ w1f,
                                                 const float* __restrict__ bias1,
                                                 bf16_t* __restrict__ h0frag,
                                                 bf16_t* __restrict__ H1,
                                                 int* __restrict__ ctrl) {
  __shared__ float  Gbuf[4][32][32];   // gate pre-activations (f,i,o,g)
  __shared__ float  cst[2][32][32];    // cell state, both layers
  __shared__ bf16_t hbuf[32][32];      // new h for frag scatter
  int* cnt = ctrl; int* phase = ctrl + 1;

  const int w = blockIdx.x;
  const int tid = threadIdx.x;
  const int lane = tid & 31, wv = tid >> 5;
  const int rt = wv & 1;
  const int g = wv >> 1;                       // gate 0..3
  const int n = lane & 15, khalf = lane >> 4;
  const int gct0 = g * 32 + 2 * w;             // global 16-col tile in gate space
  const int col0 = gct0 * 16 + n;

  // per-wave fragment base pointers (K-contiguous)
  const v16bf* wh0p0 = (const v16bf*)wh0f + (size_t)(gct0)     * 16 * 32 + lane;
  const v16bf* wh0p1 = (const v16bf*)wh0f + (size_t)(gct0 + 1) * 16 * 32 + lane;
  const v16bf* w1p0  = (const v16bf*)w1f  + (size_t)(gct0)     * 32 * 32 + lane;
  const v16bf* w1p1  = (const v16bf*)w1f  + (size_t)(gct0 + 1) * 32 * 32 + lane;

  for (int i = tid; i < 2 * 32 * 32; i += 256) (&cst[0][0][0])[i] = 0.0f;
  __syncthreads();

  int bt = 0;
  for (int s = 0; s < NS; ++s) {
    bf16_t*       h0cur  = h0frag + (size_t)(s & 1) * (2 * 16 * 512);
    const bf16_t* h0prev = h0frag + (size_t)((s & 1) ^ 1) * (2 * 16 * 512);

    // ---------------- layer 0: G0 = X0[s] + h0_prev @ Wh0 ----------------
    v8f a0, a1;
    {
      const int mb = rt * 16 + khalf * 8;
      const float* p0 = X0 + ((size_t)s * G4H + col0) * 32 + mb;
      const float* p1 = p0 + 16 * 32;
      float4 x0 = *(const float4*)p0, x1 = *(const float4*)(p0 + 4);
      float4 y0 = *(const float4*)p1, y1 = *(const float4*)(p1 + 4);
      a0[0]=x0.x; a0[1]=x0.y; a0[2]=x0.z; a0[3]=x0.w;
      a0[4]=x1.x; a0[5]=x1.y; a0[6]=x1.z; a0[7]=x1.w;
      a1[0]=y0.x; a1[1]=y0.y; a1[2]=y0.z; a1[3]=y0.w;
      a1[4]=y1.x; a1[5]=y1.y; a1[6]=y1.z; a1[7]=y1.w;
    }
    if (s > 0) {
      const v16bf* ap = (const v16bf*)h0prev + rt * 16 * 32 + lane;
      #pragma unroll
      for (int kc = 0; kc < 16; ++kc) {
        v16bf af = ap[kc * 32];
        v16bf b0 = wh0p0[kc * 32];
        v16bf b1 = wh0p1[kc * 32];
        a0 = wmma_bf16(af, b0, a0);
        a1 = wmma_bf16(af, b1, a1);
      }
    }
    {
      const int row = rt * 16 + khalf * 8;
      #pragma unroll
      for (int r = 0; r < 8; ++r) {
        Gbuf[g][row + r][n]      = a0[r];
        Gbuf[g][row + r][16 + n] = a1[r];
      }
    }
    __syncthreads();
    #pragma unroll
    for (int kk = 0; kk < 4; ++kk) {       // elementwise layer 0
      int idx = tid + kk * 256;
      int b = idx >> 5, u = idx & 31;
      float f  = sigf(Gbuf[0][b][u]);
      float ii = sigf(Gbuf[1][b][u]);
      float o  = sigf(Gbuf[2][b][u]);
      float gg = tanhf_(Gbuf[3][b][u]);
      float c = f * cst[0][b][u] + ii * gg;
      cst[0][b][u] = c;
      hbuf[b][u] = (bf16_t)(o * tanhf_(c));
    }
    __syncthreads();
    #pragma unroll
    for (int kk = 0; kk < 4; ++kk) {       // scatter h0 -> A-fragments (kc = w)
      int et = tid + kk * 256;
      int rt2 = et >> 9, r512 = et & 511;
      int lane2 = r512 >> 4, e = r512 & 15;
      int mrow = rt2 * 16 + (lane2 & 15);
      int kl = ((e >> 3) << 4) + (lane2 >> 4) * 8 + (e & 7);
      h0cur[(rt2 * 16 + w) * 512 + r512] = hbuf[mrow][kl];
    }
    grid_barrier(cnt, phase, ++bt);

    // ---------------- layer 1: G1 = bias1 + [h0_cur, h1_prev] @ W1 --------
    {
      float bb0 = bias1[col0], bb1 = bias1[col0 + 16];
      #pragma unroll
      for (int r = 0; r < 8; ++r) { a0[r] = bb0; a1[r] = bb1; }
    }
    {
      const v16bf* ap = (const v16bf*)h0cur + rt * 16 * 32 + lane;
      #pragma unroll
      for (int kc = 0; kc < 16; ++kc) {
        v16bf af = ap[kc * 32];
        v16bf b0 = w1p0[kc * 32];
        v16bf b1 = w1p1[kc * 32];
        a0 = wmma_bf16(af, b0, a0);
        a1 = wmma_bf16(af, b1, a1);
      }
    }
    if (s > 0) {
      const v16bf* ap = (const v16bf*)H1 + ((size_t)(s - 1) * 2 + rt) * 16 * 32 + lane;
      #pragma unroll
      for (int kc = 0; kc < 16; ++kc) {
        v16bf af = ap[kc * 32];
        v16bf b0 = w1p0[(16 + kc) * 32];
        v16bf b1 = w1p1[(16 + kc) * 32];
        a0 = wmma_bf16(af, b0, a0);
        a1 = wmma_bf16(af, b1, a1);
      }
    }
    {
      const int row = rt * 16 + khalf * 8;
      #pragma unroll
      for (int r = 0; r < 8; ++r) {
        Gbuf[g][row + r][n]      = a0[r];
        Gbuf[g][row + r][16 + n] = a1[r];
      }
    }
    __syncthreads();
    #pragma unroll
    for (int kk = 0; kk < 4; ++kk) {       // elementwise layer 1
      int idx = tid + kk * 256;
      int b = idx >> 5, u = idx & 31;
      float f  = sigf(Gbuf[0][b][u]);
      float ii = sigf(Gbuf[1][b][u]);
      float o  = sigf(Gbuf[2][b][u]);
      float gg = tanhf_(Gbuf[3][b][u]);
      float c = f * cst[1][b][u] + ii * gg;
      cst[1][b][u] = c;
      hbuf[b][u] = (bf16_t)(o * tanhf_(c));
    }
    __syncthreads();
    #pragma unroll
    for (int kk = 0; kk < 4; ++kk) {       // scatter h1 -> H1[s] A-fragments
      int et = tid + kk * 256;
      int rt2 = et >> 9, r512 = et & 511;
      int lane2 = r512 >> 4, e = r512 & 15;
      int mrow = rt2 * 16 + (lane2 & 15);
      int kl = ((e >> 3) << 4) + (lane2 >> 4) * 8 + (e & 7);
      H1[(((size_t)s * 2 + rt2) * 16 + w) * 512 + r512] = hbuf[mrow][kl];
    }
    grid_barrier(cnt, phase, ++bt);
  }
}

// ---------------------------------------------------------------------------
// Decoder: out[b,s,:] = h1[s,b] @ W_dec + b_dec.
// grid = (20, 256), 256 threads (8 waves). H1[s] staged in LDS once per block;
// each wave: 4 col tiles x 2 row tiles = 8 accumulators, K=512 (16 kc).
// Per kc: 4 global B loads + 2 LDS A loads for 8 WMMAs.
// ---------------------------------------------------------------------------
__global__ __launch_bounds__(256) void k_dec(const bf16_t* __restrict__ H1,
                                             const bf16_t* __restrict__ wdf,
                                             const float* __restrict__ b_dec,
                                             float* __restrict__ out) {
  __shared__ bf16_t Ash[2 * 16 * 512];               // H1[s] A-fragments (32 KB)
  const int s = blockIdx.y;
  const int tid = threadIdx.x;
  const int wv = tid >> 5, lane = tid & 31;

  {                                                  // cooperative stage of H1[s]
    const uint4* src = (const uint4*)(H1 + (size_t)s * 2 * 16 * 512);
    uint4* dstp = (uint4*)Ash;
    #pragma unroll
    for (int i = 0; i < 8; ++i) dstp[tid + i * 256] = src[tid + i * 256];
  }
  __syncthreads();

  const int ct0 = (blockIdx.x * 8 + wv) * 4;         // 4 col tiles per wave
  if (ct0 < DEC_CT) {
    const int n = lane & 15, khalf = lane >> 4;
    int ct[4];
    bool val[4];
    #pragma unroll
    for (int i = 0; i < 4; ++i) {
      val[i] = (ct0 + i) < DEC_CT;
      ct[i] = val[i] ? (ct0 + i) : (DEC_CT - 1);     // clamp for loads
    }
    const v16bf* bp[4];
    #pragma unroll
    for (int i = 0; i < 4; ++i)
      bp[i] = (const v16bf*)wdf + (size_t)ct[i] * 16 * 32 + lane;

    v8f acc[4][2];
    #pragma unroll
    for (int i = 0; i < 4; ++i) {
      float bb = b_dec[ct[i] * 16 + n];
      #pragma unroll
      for (int r = 0; r < 8; ++r) { acc[i][0][r] = bb; acc[i][1][r] = bb; }
    }

    #pragma unroll 4
    for (int kc = 0; kc < 16; ++kc) {
      v16bf af0 = *(const v16bf*)&Ash[(0 * 16 + kc) * 512 + lane * 16];
      v16bf af1 = *(const v16bf*)&Ash[(1 * 16 + kc) * 512 + lane * 16];
      #pragma unroll
      for (int i = 0; i < 4; ++i) {
        if (kc + 4 < 16) __builtin_prefetch((const void*)(bp[i] + (kc + 4) * 32), 0, 1);
        v16bf b = bp[i][kc * 32];
        acc[i][0] = wmma_bf16(af0, b, acc[i][0]);
        acc[i][1] = wmma_bf16(af1, b, acc[i][1]);
      }
    }

    #pragma unroll
    for (int i = 0; i < 4; ++i) {
      if (!val[i]) continue;
      const int col = ct[i] * 16 + n;
      #pragma unroll
      for (int r = 0; r < 8; ++r) {
        int brow = khalf * 8 + r;
        out[((size_t)brow * NS + s) * NO + col]        = acc[i][0][r];
        out[((size_t)(brow + 16) * NS + s) * NO + col] = acc[i][1][r];
      }
    }
  }
}

// ---------------------------------------------------------------------------
extern "C" void kernel_launch(void* const* d_in, const int* in_sizes, int n_in,
                              void* d_out, int out_size, void* d_ws, size_t ws_size,
                              hipStream_t stream) {
  const int*   ids  = (const int*)d_in[0];
  const float* emb  = (const float*)d_in[1];
  const float* W_i  = (const float*)d_in[2];
  const float* b_i  = (const float*)d_in[3];
  const float* W_h  = (const float*)d_in[4];
  const float* b_h  = (const float*)d_in[5];
  const float* Wdec = (const float*)d_in[6];
  const float* bdec = (const float*)d_in[7];
  float* out = (float*)d_out;
  (void)in_sizes; (void)n_in; (void)out_size; (void)ws_size;

  char* ws = (char*)d_ws;
  size_t off = 0;
  auto carve = [&](size_t bytes) -> void* {
    off = (off + 255) & ~(size_t)255;
    void* p = ws + off; off += bytes; return p;
  };
  int*    ctrl   = (int*)   carve(256);
  float*  bias1  = (float*) carve((size_t)2048 * 4);
  bf16_t* wi0f   = (bf16_t*)carve((size_t)128 * 16 * 512 * 2);
  bf16_t* wh0f   = (bf16_t*)carve((size_t)128 * 16 * 512 * 2);
  bf16_t* w1f    = (bf16_t*)carve((size_t)128 * 32 * 512 * 2);
  bf16_t* wdf    = (bf16_t*)carve((size_t)DEC_CT * 16 * 512 * 2);
  float*  X0     = (float*) carve((size_t)NS * G4H * 32 * 4);
  bf16_t* H1     = (bf16_t*)carve((size_t)NS * 2 * 16 * 512 * 2);
  bf16_t* h0frag = (bf16_t*)carve((size_t)2 * 2 * 16 * 512 * 2);

  hipMemsetAsync(ctrl, 0, 256, stream);
  k_repack_gate512<<<4096, 256, 0, stream>>>(W_i, wi0f);
  k_repack_gate512<<<4096, 256, 0, stream>>>(W_h, wh0f);
  k_repack_w1<<<8192, 256, 0, stream>>>(W_i + (size_t)4 * 512 * 512,
                                        W_h + (size_t)4 * 512 * 512, w1f);
  k_repack_wdec<<<20000, 256, 0, stream>>>(Wdec, wdf);
  k_bias1<<<8, 256, 0, stream>>>(b_i, b_h, bias1);
  k_x0<<<dim3(16, 256), 256, 0, stream>>>(ids, emb, b_i, b_h, wi0f, X0);
  k_lstm<<<NWG, 256, 0, stream>>>(X0, wh0f, w1f, bias1, h0frag, H1, ctrl);
  k_dec<<<dim3(20, 256), 256, 0, stream>>>(H1, wdf, bdec, out);
}
